// DeepseekV2Attention_472446403327
// MI455X (gfx1250) — matrix operand order repaired and checked by reading, not compiled, not run on hardware
//
#include <hip/hip_runtime.h>

// ---------------------------------------------------------------------------
// Types
// ---------------------------------------------------------------------------
typedef __bf16 bf16;
typedef __attribute__((ext_vector_type(16))) __bf16 v16bf;
typedef __attribute__((ext_vector_type(8)))  __bf16 v8bf;
typedef __attribute__((ext_vector_type(8)))  float  v8f;

__device__ __forceinline__ v8f wmma_bf16(v16bf a, v16bf b, v8f c) {
  // D = A(16x32) * B(32x16) + C, f32 accumulate
  return __builtin_amdgcn_wmma_f32_16x16x32_bf16(
      /*neg_a=*/false, a, /*neg_b=*/false, b,
      /*c_mod=*/(short)0, c, /*reuse_a=*/false, /*reuse_b=*/false);
}

__device__ __forceinline__ v16bf pack16(v8bf lo, v8bf hi) {
  v16bf r;
#pragma unroll
  for (int e = 0; e < 8; ++e) { r[e] = lo[e]; r[e + 8] = hi[e]; }
  return r;
}

// ---------------------------------------------------------------------------
// Problem constants
// ---------------------------------------------------------------------------
#define CB    2
#define CS    2048
#define CHID  4096
#define CH    32
#define CNOPE 128
#define CROPE 64
#define CVD   128
#define CQK   192
#define CQLR  1536
#define CKVLR 512

// ---------------------------------------------------------------------------
// fp32 -> bf16 elementwise convert
// ---------------------------------------------------------------------------
__global__ __launch_bounds__(256) void cvt_f32_bf16(
    const float* __restrict__ x, bf16* __restrict__ y, int n) {
  int i = blockIdx.x * 256 + threadIdx.x;
  if (i < n) y[i] = (bf16)x[i];
}

// ---------------------------------------------------------------------------
// RMSNorm: one block per row, bf16 in -> bf16 out (f32 math)
// ---------------------------------------------------------------------------
__global__ __launch_bounds__(256) void rmsnorm_bf16(
    const bf16* __restrict__ x, const float* __restrict__ w,
    bf16* __restrict__ y, int D, int in_stride, int out_stride) {
  const int row = blockIdx.x;
  const bf16* xr = x + (size_t)row * in_stride;
  bf16* yr = y + (size_t)row * out_stride;
  float ss = 0.f;
  for (int i = threadIdx.x; i < D; i += 256) {
    float v = (float)xr[i];
    ss += v * v;
  }
  __shared__ float red[256];
  red[threadIdx.x] = ss;
  __syncthreads();
  for (int off = 128; off > 0; off >>= 1) {
    if (threadIdx.x < off) red[threadIdx.x] += red[threadIdx.x + off];
    __syncthreads();
  }
  const float rstd = rsqrtf(red[0] / (float)D + 1e-6f);
  for (int i = threadIdx.x; i < D; i += 256)
    yr[i] = (bf16)((float)xr[i] * rstd * w[i]);
}

// ---------------------------------------------------------------------------
// Tiled bf16 WMMA GEMM:  C[M,N] = A[M,K] * W[N,K]^T
// Block tile 128x64, 8 waves (4x2), wave tile 32x32 (2x2 WMMA tiles).
// Global->LDS staging is software-pipelined through registers so HBM/L2
// latency overlaps the WMMA compute phase.
// Requires M%128==0, N%64==0, K%32==0 (true for all calls here).
// ---------------------------------------------------------------------------
template <typename OutT>
__global__ __launch_bounds__(256) void gemm_bf16_tn(
    const bf16* __restrict__ A, const bf16* __restrict__ W,
    OutT* __restrict__ C, int M, int N, int K) {
  __shared__ __align__(16) bf16 Asm[128][40];  // pad rows to 80B (16B aligned)
  __shared__ __align__(16) bf16 Wsm[64][40];

  const int tid  = threadIdx.x;
  const int wave = tid >> 5;
  const int lane = tid & 31;
  const int half = lane >> 4;
  const int l16  = lane & 15;
  const int wm   = wave >> 1;  // 0..3
  const int wn   = wave & 1;   // 0..1
  const int m0   = blockIdx.y * 128;
  const int n0   = blockIdx.x * 64;

  // Staging assignments (per thread): A rows a_r, a_r+64; W row w_r.
  const int a_r = tid >> 2;
  const int a_c = (tid & 3) << 3;
  const bf16* Arow0 = A + (size_t)(m0 + a_r) * K + a_c;
  const bf16* Arow1 = A + (size_t)(m0 + a_r + 64) * K + a_c;
  const bf16* Wrow  = W + (size_t)(n0 + a_r) * K + a_c;  // a_r<64 always

  v8bf pre_a0, pre_a1, pre_w;

  // Prologue: fetch + stage tile 0
  pre_a0 = *(const v8bf*)(Arow0);
  pre_a1 = *(const v8bf*)(Arow1);
  pre_w  = *(const v8bf*)(Wrow);
  *(v8bf*)&Asm[a_r][a_c]      = pre_a0;
  *(v8bf*)&Asm[a_r + 64][a_c] = pre_a1;
  *(v8bf*)&Wsm[a_r][a_c]      = pre_w;

  v8f acc[2][2];
#pragma unroll
  for (int i = 0; i < 2; ++i)
#pragma unroll
    for (int j = 0; j < 2; ++j) acc[i][j] = (v8f)(0.f);

  for (int k0 = 0; k0 < K; k0 += 32) {
    __syncthreads();  // staged tile visible to all waves
    const bool more = (k0 + 32) < K;
    if (more) {  // prefetch next slab while we compute on this one
      pre_a0 = *(const v8bf*)(Arow0 + k0 + 32);
      pre_a1 = *(const v8bf*)(Arow1 + k0 + 32);
      pre_w  = *(const v8bf*)(Wrow + k0 + 32);
    }

    // A fragments: lane l16 = row M; K = half*8+e (lo), 16+half*8+e (hi)
    v16bf af[2];
#pragma unroll
    for (int i = 0; i < 2; ++i) {
      int m = wm * 32 + i * 16 + l16;
      v8bf lo = *(const v8bf*)&Asm[m][half * 8];
      v8bf hi = *(const v8bf*)&Asm[m][16 + half * 8];
      af[i] = pack16(lo, hi);
    }
    // B fragments: lane l16 = col N; K = half*8+e (lo), 16+half*8+e (hi)
    v16bf bfg[2];
#pragma unroll
    for (int j = 0; j < 2; ++j) {
      int n = wn * 32 + j * 16 + l16;
      v8bf lo = *(const v8bf*)&Wsm[n][half * 8];
      v8bf hi = *(const v8bf*)&Wsm[n][16 + half * 8];
      bfg[j] = pack16(lo, hi);
    }
#pragma unroll
    for (int i = 0; i < 2; ++i)
#pragma unroll
      for (int j = 0; j < 2; ++j)
        acc[i][j] = wmma_bf16(af[i], bfg[j], acc[i][j]);

    __syncthreads();  // all waves done reading LDS
    if (more) {
      *(v8bf*)&Asm[a_r][a_c]      = pre_a0;
      *(v8bf*)&Asm[a_r + 64][a_c] = pre_a1;
      *(v8bf*)&Wsm[a_r][a_c]      = pre_w;
    }
  }

  // Epilogue: C layout -> (m = base + half*8 + r, n = base + l16)
#pragma unroll
  for (int i = 0; i < 2; ++i)
#pragma unroll
    for (int j = 0; j < 2; ++j) {
      int n = n0 + wn * 32 + j * 16 + l16;
#pragma unroll
      for (int r = 0; r < 8; ++r) {
        int m = m0 + wm * 32 + i * 16 + half * 8 + r;
        C[(size_t)m * N + n] = (OutT)acc[i][j][r];
      }
    }
}

// ---------------------------------------------------------------------------
// RoPE + pack Q: qb (B,S,H*192) bf16 -> Q (B,H,S,192) bf16
// ---------------------------------------------------------------------------
__global__ __launch_bounds__(256) void rope_q_pack(
    const bf16* __restrict__ qb, const float* __restrict__ cosb,
    const float* __restrict__ sinb, bf16* __restrict__ Qout) {
  int idx = blockIdx.x * 256 + threadIdx.x;  // over B*S*H
  if (idx >= CB * CS * CH) return;
  int h = idx & (CH - 1);
  int s = (idx >> 5) & (CS - 1);
  int b = idx >> 16;  // idx / (S*H), S*H = 65536
  const bf16* src = qb + (size_t)(b * CS + s) * (CH * CQK) + h * CQK;
  const float* cp = cosb + (size_t)(b * CS + s) * (CROPE / 2);
  const float* sp = sinb + (size_t)(b * CS + s) * (CROPE / 2);
  bf16* dst = Qout + (((size_t)b * CH + h) * CS + s) * CQK;
#pragma unroll
  for (int d = 0; d < CNOPE; d += 8)
    *(v8bf*)&dst[d] = *(const v8bf*)&src[d];
#pragma unroll
  for (int i = 0; i < CROPE / 2; ++i) {
    float xr = (float)src[CNOPE + 2 * i];
    float xi = (float)src[CNOPE + 2 * i + 1];
    float c = cp[i], sn = sp[i];
    dst[CNOPE + 2 * i]     = (bf16)(xr * c - xi * sn);
    dst[CNOPE + 2 * i + 1] = (bf16)(xr * sn + xi * c);
  }
}

// ---------------------------------------------------------------------------
// RoPE + pack K/V: kvb (B,S,H*256), ckv (B,S,576) -> K (B,H,S,192), V (B,H,S,128)
// ---------------------------------------------------------------------------
__global__ __launch_bounds__(256) void rope_kv_pack(
    const bf16* __restrict__ kvb, const bf16* __restrict__ ckv,
    const float* __restrict__ cosb, const float* __restrict__ sinb,
    bf16* __restrict__ Kout, bf16* __restrict__ Vout) {
  int idx = blockIdx.x * 256 + threadIdx.x;
  if (idx >= CB * CS * CH) return;
  int h = idx & (CH - 1);
  int s = (idx >> 5) & (CS - 1);
  int b = idx >> 16;
  const bf16* src = kvb + (size_t)(b * CS + s) * (CH * (CNOPE + CVD)) + h * (CNOPE + CVD);
  const bf16* kr  = ckv + (size_t)(b * CS + s) * (CKVLR + CROPE) + CKVLR;
  const float* cp = cosb + (size_t)(b * CS + s) * (CROPE / 2);
  const float* sp = sinb + (size_t)(b * CS + s) * (CROPE / 2);
  bf16* dstK = Kout + (((size_t)b * CH + h) * CS + s) * CQK;
  bf16* dstV = Vout + (((size_t)b * CH + h) * CS + s) * CVD;
#pragma unroll
  for (int d = 0; d < CNOPE; d += 8) {
    *(v8bf*)&dstK[d] = *(const v8bf*)&src[d];
    *(v8bf*)&dstV[d] = *(const v8bf*)&src[CNOPE + d];
  }
#pragma unroll
  for (int i = 0; i < CROPE / 2; ++i) {
    float xr = (float)kr[2 * i];
    float xi = (float)kr[2 * i + 1];
    float c = cp[i], sn = sp[i];
    dstK[CNOPE + 2 * i]     = (bf16)(xr * c - xi * sn);
    dstK[CNOPE + 2 * i + 1] = (bf16)(xr * sn + xi * c);
  }
}

// ---------------------------------------------------------------------------
// Flash attention (causal), bf16 WMMA, f32 online softmax.
// grid = (S/128, H, B), block = 256 (8 waves). Wave w: 16 query rows.
// Q,K: (B,H,S,192); V: (B,H,S,128). Out: (B,S,H*128) bf16.
// Fragment loads are batched ahead of each WMMA group so a single
// s_wait_dscnt covers 8 ds_load_b128 and WMMAs issue back-to-back.
// ---------------------------------------------------------------------------
__global__ __launch_bounds__(256) void mla_flash_attn(
    const bf16* __restrict__ Q, const bf16* __restrict__ Kmat,
    const bf16* __restrict__ Vmat, bf16* __restrict__ Oattn) {
  const int qt = blockIdx.x;
  const int h  = blockIdx.y;
  const int b  = blockIdx.z;

  __shared__ __align__(16) bf16 Ksm[64][200];   // 64 keys x 192 (+pad)
  __shared__ __align__(16) bf16 VTsm[128][72];  // transposed: [dim][key] (+pad)
  __shared__ __align__(16) bf16 Psm[8][16][64]; // per-wave P tile

  const int tid  = threadIdx.x;
  const int wave = tid >> 5;
  const int lane = tid & 31;
  const int half = lane >> 4;
  const int l16  = lane & 15;
  const int qbase = qt * 128 + wave * 16;
  const size_t bh = (size_t)b * CH + h;

  // Preload Q fragments (16 rows x 192, 6 K-chunks of 32)
  const bf16* Qp = Q + (bh * CS + qbase) * CQK;
  v16bf qf[6];
#pragma unroll
  for (int kc = 0; kc < 6; ++kc) {
    const bf16* row = Qp + (size_t)l16 * CQK + kc * 32;
    v8bf lo = *(const v8bf*)(row + half * 8);
    v8bf hi = *(const v8bf*)(row + 16 + half * 8);
    qf[kc] = pack16(lo, hi);
  }

  v8f ao[8];
#pragma unroll
  for (int j = 0; j < 8; ++j) ao[j] = (v8f)(0.f);
  float mrow[8], lrow[8];
#pragma unroll
  for (int r = 0; r < 8; ++r) { mrow[r] = -1e30f; lrow[r] = 0.f; }

  const float scal = 0.07216878364870323f;  // 1/sqrt(192)
  const int nkt = (qt + 1) * 2;

  for (int kt = 0; kt < nkt; ++kt) {
    // --- Stage K tile: 64x192 bf16 (1536 16B chunks; 6/thread) ---
    const bf16* Kp = Kmat + (bh * CS + (size_t)kt * 64) * CQK;
#pragma unroll
    for (int i = 0; i < 6; ++i) {
      int c  = tid + i * 256;
      int r  = c / 24;
      int cc = (c % 24) * 8;
      *(v8bf*)&Ksm[r][cc] = *(const v8bf*)&Kp[(size_t)r * CQK + cc];
    }
    // --- Stage V tile transposed: 32x128 (keys 0..31 only) -> VTsm[dim][key] ---
    const bf16* Vp = Vmat + (bh * CS + (size_t)kt * 64) * CVD;
#pragma unroll
    for (int i = 0; i < 2; ++i) {
      int c   = tid + i * 256;
      int key = c >> 4;
      int d8  = (c & 15) * 8;
      v8bf v = *(const v8bf*)&Vp[(size_t)key * CVD + d8];
#pragma unroll
      for (int e = 0; e < 8; ++e) VTsm[d8 + e][key] = v[e];
    }
    __syncthreads();

    if (kt * 64 <= qbase) {  // skip tiles fully above the diagonal
      // --- Scores: 16x64 = 4 N-tiles, accumulate over 6 K-chunks ---
      v8f sc[4];
#pragma unroll
      for (int j = 0; j < 4; ++j) sc[j] = (v8f)(0.f);
#pragma unroll
      for (int kc = 0; kc < 6; ++kc) {
        v16bf bk[4];  // batch all 4 B fragments, then 4 WMMAs back-to-back
#pragma unroll
        for (int j = 0; j < 4; ++j) {
          int n = j * 16 + l16;  // key row in tile
          v8bf lo = *(const v8bf*)&Ksm[n][kc * 32 + half * 8];
          v8bf hi = *(const v8bf*)&Ksm[n][kc * 32 + 16 + half * 8];
          bk[j] = pack16(lo, hi);
        }
#pragma unroll
        for (int j = 0; j < 4; ++j)
          sc[j] = wmma_bf16(qf[kc], bk[j], sc[j]);
      }
      // --- Scale + causal mask ---
      const int kb = kt * 64;
#pragma unroll
      for (int j = 0; j < 4; ++j) {
        int k_abs = kb + j * 16 + l16;
#pragma unroll
        for (int r = 0; r < 8; ++r) {
          int q_abs = qbase + half * 8 + r;
          float s = sc[j][r] * scal;
          sc[j][r] = (k_abs > q_abs) ? -1e30f : s;
        }
      }
      // --- Online softmax per row (width-16 xor-shuffle reductions) ---
#pragma unroll
      for (int r = 0; r < 8; ++r) {
        float mx = sc[0][r];
#pragma unroll
        for (int j = 1; j < 4; ++j) mx = fmaxf(mx, sc[j][r]);
#pragma unroll
        for (int msk = 1; msk < 16; msk <<= 1)
          mx = fmaxf(mx, __shfl_xor(mx, msk, 32));
        float mnew = fmaxf(mrow[r], mx);
        float corr = __expf(mrow[r] - mnew);
        float pj[4], psum = 0.f;
#pragma unroll
        for (int j = 0; j < 4; ++j) { pj[j] = __expf(sc[j][r] - mnew); psum += pj[j]; }
#pragma unroll
        for (int msk = 1; msk < 16; msk <<= 1)
          psum += __shfl_xor(psum, msk, 32);
        mrow[r] = mnew;
        lrow[r] = lrow[r] * corr + psum;
#pragma unroll
        for (int j2 = 0; j2 < 8; ++j2) ao[j2][r] *= corr;
#pragma unroll
        for (int j = 0; j < 4; ++j)
          Psm[wave][half * 8 + r][j * 16 + l16] = (bf16)pj[j];
      }
      // --- PV: O(16x128) += P(16x64) * V(64x128) ---
#pragma unroll
      for (int kc2 = 0; kc2 < 2; ++kc2) {
        v8bf plo = *(const v8bf*)&Psm[wave][l16][kc2 * 32 + half * 8];
        v8bf phi = *(const v8bf*)&Psm[wave][l16][kc2 * 32 + 16 + half * 8];
        v16bf pa = pack16(plo, phi);
#pragma unroll
        for (int g = 0; g < 2; ++g) {  // groups of 4 to bound VGPR pressure
          v16bf vb[4];
#pragma unroll
          for (int jj = 0; jj < 4; ++jj) {
            int n = (g * 4 + jj) * 16 + l16;  // output dim
            v8bf lo = *(const v8bf*)&VTsm[n][kc2 * 32 + half * 8];
            v8bf hi = *(const v8bf*)&VTsm[n][kc2 * 32 + 16 + half * 8];
            vb[jj] = pack16(lo, hi);
          }
#pragma unroll
          for (int jj = 0; jj < 4; ++jj)
            ao[g * 4 + jj] = wmma_bf16(pa, vb[jj], ao[g * 4 + jj]);
        }
      }
    }
    __syncthreads();
  }

  // --- Epilogue: normalize, store to (B,S,H*128) bf16 ---
#pragma unroll
  for (int j2 = 0; j2 < 8; ++j2) {
    int n = h * CVD + j2 * 16 + l16;
#pragma unroll
    for (int r = 0; r < 8; ++r) {
      int q_abs = qbase + half * 8 + r;
      float o = ao[j2][r] / lrow[r];
      Oattn[((size_t)b * CS + q_abs) * (CH * CVD) + n] = (bf16)o;
    }
  }
}

// ---------------------------------------------------------------------------
// Host orchestration
// ---------------------------------------------------------------------------
extern "C" void kernel_launch(void* const* d_in, const int* in_sizes, int n_in,
                              void* d_out, int out_size, void* d_ws, size_t ws_size,
                              hipStream_t stream) {
  (void)in_sizes; (void)n_in; (void)out_size; (void)ws_size;
  const float* hidden    = (const float*)d_in[0];
  const float* cosb      = (const float*)d_in[1];
  const float* sinb      = (const float*)d_in[2];
  const float* q_a_w     = (const float*)d_in[3];
  const float* q_a_ln_w  = (const float*)d_in[4];
  const float* q_b_w     = (const float*)d_in[5];
  const float* kv_a_w    = (const float*)d_in[6];
  const float* kv_a_ln_w = (const float*)d_in[7];
  const float* kv_b_w    = (const float*)d_in[8];
  const float* o_w       = (const float*)d_in[9];
  float* out = (float*)d_out;

  const int M = CB * CS;  // 4096 token rows

  char* ws = (char*)d_ws;
  size_t off = 0;
  auto alloc = [&](size_t bytes) -> void* {
    void* p = ws + off;
    off = (off + bytes + 255) & ~(size_t)255;
    return p;
  };

  bf16* hid_bf   = (bf16*)alloc((size_t)M * CHID * 2);
  bf16* wqa_bf   = (bf16*)alloc((size_t)CQLR * CHID * 2);
  bf16* wkva_bf  = (bf16*)alloc((size_t)(CKVLR + CROPE) * CHID * 2);
  bf16* wqb_bf   = (bf16*)alloc((size_t)(CH * CQK) * CQLR * 2);
  bf16* wkvb_bf  = (bf16*)alloc((size_t)(CH * (CNOPE + CVD)) * CKVLR * 2);
  bf16* wo_bf    = (bf16*)alloc((size_t)CHID * (CH * CVD) * 2);
  bf16* qa_bf    = (bf16*)alloc((size_t)M * CQLR * 2);
  bf16* qa_nrm   = (bf16*)alloc((size_t)M * CQLR * 2);
  bf16* ckv_bf   = (bf16*)alloc((size_t)M * (CKVLR + CROPE) * 2);
  bf16* ckv_nrm  = (bf16*)alloc((size_t)M * CKVLR * 2);
  bf16* qb_bf    = (bf16*)alloc((size_t)M * (CH * CQK) * 2);
  bf16* kvb_bf   = (bf16*)alloc((size_t)M * (CH * (CNOPE + CVD)) * 2);
  bf16* Qattn    = (bf16*)alloc((size_t)CB * CH * CS * CQK * 2);
  bf16* Kattn    = (bf16*)alloc((size_t)CB * CH * CS * CQK * 2);
  bf16* Vattn    = (bf16*)alloc((size_t)CB * CH * CS * CVD * 2);
  bf16* attn_bf  = (bf16*)alloc((size_t)M * (CH * CVD) * 2);

  auto cvt = [&](const float* src, bf16* dst, int n) {
    cvt_f32_bf16<<<(n + 255) / 256, 256, 0, stream>>>(src, dst, n);
  };
  cvt(hidden, hid_bf,  M * CHID);
  cvt(q_a_w,  wqa_bf,  CQLR * CHID);
  cvt(kv_a_w, wkva_bf, (CKVLR + CROPE) * CHID);
  cvt(q_b_w,  wqb_bf,  CH * CQK * CQLR);
  cvt(kv_b_w, wkvb_bf, CH * (CNOPE + CVD) * CKVLR);
  cvt(o_w,    wo_bf,   CHID * CH * CVD);

  // q_a = hidden @ q_a_w^T ; ckv = hidden @ kv_a_w^T
  gemm_bf16_tn<bf16><<<dim3(CQLR / 64, M / 128), 256, 0, stream>>>(
      hid_bf, wqa_bf, qa_bf, M, CQLR, CHID);
  gemm_bf16_tn<bf16><<<dim3((CKVLR + CROPE) / 64, M / 128), 256, 0, stream>>>(
      hid_bf, wkva_bf, ckv_bf, M, CKVLR + CROPE, CHID);

  // RMSNorms
  rmsnorm_bf16<<<M, 256, 0, stream>>>(qa_bf, q_a_ln_w, qa_nrm,
                                      CQLR, CQLR, CQLR);
  rmsnorm_bf16<<<M, 256, 0, stream>>>(ckv_bf, kv_a_ln_w, ckv_nrm,
                                      CKVLR, CKVLR + CROPE, CKVLR);

  // q = rms(q_a) @ q_b_w^T ; kv = rms(ckv) @ kv_b_w^T
  gemm_bf16_tn<bf16><<<dim3(CH * CQK / 64, M / 128), 256, 0, stream>>>(
      qa_nrm, wqb_bf, qb_bf, M, CH * CQK, CQLR);
  gemm_bf16_tn<bf16><<<dim3(CH * (CNOPE + CVD) / 64, M / 128), 256, 0, stream>>>(
      ckv_nrm, wkvb_bf, kvb_bf, M, CH * (CNOPE + CVD), CKVLR);

  // RoPE + repack into attention layouts
  const int nbh = CB * CS * CH;
  rope_q_pack<<<(nbh + 255) / 256, 256, 0, stream>>>(qb_bf, cosb, sinb, Qattn);
  rope_kv_pack<<<(nbh + 255) / 256, 256, 0, stream>>>(kvb_bf, ckv_bf, cosb, sinb,
                                                      Kattn, Vattn);

  // Flash attention
  mla_flash_attn<<<dim3(CS / 128, CH, CB), 256, 0, stream>>>(Qattn, Kattn, Vattn,
                                                             attn_bf);

  // out = attn @ o_w^T (f32 output)
  gemm_bf16_tn<float><<<dim3(CHID / 64, M / 128), 256, 0, stream>>>(
      attn_bf, wo_bf, out, M, CHID, CH * CVD);
}
